// BERT_CrossAttention_Model_62740882260354
// MI455X (gfx1250) — compile-verified, matrix operand order checked by reading, not run on hardware
//
#include <hip/hip_runtime.h>

// ---------------------------------------------------------------------------
// CDNA5 / gfx1250 implementation of the BERT cross-attention model.
// wave32, WMMA bf16 16x16x32 (f32 accumulate), Tensor-Data-Mover (TDM)
// staged GEMM tiles with LDS double buffering, flash-attention w/ online
// softmax.
// ---------------------------------------------------------------------------

typedef __bf16 bf16_t;
typedef __attribute__((ext_vector_type(16))) __bf16 v16bf;
typedef __attribute__((ext_vector_type(8)))  __bf16 v8bf;
typedef __attribute__((ext_vector_type(8)))  float  v8f;
typedef __attribute__((ext_vector_type(4))) unsigned int v4u;
typedef __attribute__((ext_vector_type(8))) int v8i;
typedef __attribute__((ext_vector_type(4))) int v4i;

#define WMMA_BF16(a, b, c) \
  __builtin_amdgcn_wmma_f32_16x16x32_bf16(false, (a), false, (b), (short)0, (c), false, false)

// LDS byte offset of a shared-memory pointer: flat LDS addresses keep the
// offset in the low 32 bits (ISA 10.2 aperture rules), so an addrspace(3)
// cast + truncation yields the dsaddr the TDM descriptor wants.
typedef __attribute__((address_space(3))) const void lds_cv;
__device__ __forceinline__ unsigned lds_off(const void* p) {
  return (unsigned)(unsigned long long)(lds_cv*)p;
}

// ---------------------------------------------------------------------------
// TDM: DMA a rows x cols bf16 tile (row stride = strideElems) from global
// memory into LDS, inserting 16B of padding after each row (cols = 64 elems
// = 32 DWORDs -> pad_interval code 4; 16B = 4 DWORDs -> pad_amount code 3),
// so the LDS tile has ld = cols + 8 elements (bank-conflict-free for the
// WMMA fragment loader). Descriptor layout per CDNA5 ISA 8.3 / 8.4.
// This toolchain exposes the 6-argument builtin:
//   (uint32x4 g0, int32x8 g1, int32x4 g2, int32x4 g3, int32x8 g4, i32 cpol)
// g2/g3/g4 are only needed for >2D tensors / gather mode -> zero-filled here.
// ---------------------------------------------------------------------------
__device__ __forceinline__ void tdm_load_2d(const bf16_t* gptr, unsigned ldsByteOff,
                                            int cols, int rows, int strideElems,
                                            int tensorRows) {
  unsigned long long ga = (unsigned long long)gptr;
  v4u g0;
  g0[0] = 1u;                                 // count=1 (valid user descriptor)
  g0[1] = ldsByteOff;                         // lds_addr
  g0[2] = (unsigned)(ga & 0xffffffffu);       // global_addr[31:0]
  g0[3] = (unsigned)(ga >> 32) | (2u << 30);  // global_addr[56:32] | type=2
  v8i g1;
  g1[0] = (1 << 16)            // data_size = 2 bytes
        | (1 << 20)            // pad_enable
        | (4 << 22)            // pad_interval: 2^(4+1) = 32 DWORDs (= 64 bf16)
        | (3 << 25);           // pad_amount: 3+1 = 4 DWORDs (= 8 bf16)
  g1[1] = (strideElems & 0xffff) << 16;                         // tensor_dim0 lo
  g1[2] = ((strideElems >> 16) & 0xffff) | ((tensorRows & 0xffff) << 16);
  g1[3] = ((tensorRows >> 16) & 0xffff) | ((cols & 0xffff) << 16); // tile_dim0
  g1[4] = rows & 0xffff;       // tile_dim1 (tile_dim2 = 0 -> 2D tile)
  g1[5] = strideElems;         // tensor_dim0_stride[31:0]
  g1[6] = 0;                   // stride hi | tensor_dim1_stride lo (unused, 2D)
  g1[7] = 0;
  v4i g2 = {0, 0, 0, 0};
  v4i g3 = {0, 0, 0, 0};
  v8i g4 = {0, 0, 0, 0, 0, 0, 0, 0};
  __builtin_amdgcn_tensor_load_to_lds(g0, g1, g2, g3, g4, 0);
}

// Load a 16x32 bf16 A-fragment (or B^T-fragment) from an LDS tile.
// CDNA5 layout: lane m = lane&15 owns row m; K base = (lane>>4)*8;
// vector elems 0..7 = K[kb..kb+7], elems 8..15 = K[kb+16..kb+23].
// ld must be a multiple of 8 elements so the two 16B loads stay aligned.
__device__ __forceinline__ v16bf load_frag(const bf16_t* base, int ld, int row0,
                                           int k0, int lane) {
  const bf16_t* p = base + (row0 + (lane & 15)) * ld + k0 + ((lane >> 4) << 3);
  v8bf lo = *(const v8bf*)(p);
  v8bf hi = *(const v8bf*)(p + 16);
  return __builtin_shufflevector(lo, hi, 0,1,2,3,4,5,6,7,8,9,10,11,12,13,14,15);
}

// ---------------------------------------------------------------------------
// Weight transpose + fp32 -> bf16 conversion: Wt[n][k] = (bf16)W[k][n], D=1024
// ---------------------------------------------------------------------------
__global__ void __launch_bounds__(256) transpose_convert_kernel(
    const float* __restrict__ W, bf16_t* __restrict__ Wt) {
  __shared__ float tile[32][33];
  const int bx = blockIdx.x * 32;   // n
  const int by = blockIdx.y * 32;   // k
  const int tx = threadIdx.x, ty = threadIdx.y;
#pragma unroll
  for (int i = 0; i < 4; ++i)
    tile[ty + i * 8][tx] = W[(size_t)(by + ty + i * 8) * 1024 + bx + tx];
  __syncthreads();
#pragma unroll
  for (int i = 0; i < 4; ++i)
    Wt[(size_t)(bx + ty + i * 8) * 1024 + by + tx] = (bf16_t)tile[tx][ty + i * 8];
}

// ---------------------------------------------------------------------------
// Split embedded into premise/hypothesis halves, fp32 -> bf16.
// dst[(b*512+s)*1024 + d] = emb[(b*1024 + off + s)*1024 + d]
// ---------------------------------------------------------------------------
__global__ void __launch_bounds__(256) split_bf16_kernel(
    const float* __restrict__ emb, bf16_t* __restrict__ dst, int off) {
  size_t idx = ((size_t)blockIdx.x * 256 + threadIdx.x) * 4;
  int row = (int)(idx >> 10);
  int d   = (int)(idx & 1023);
  int b = row >> 9, s = row & 511;
  float4 v = *(const float4*)(emb + (((size_t)(b * 1024 + off + s)) << 10) + d);
  bf16_t* o = dst + idx;
  o[0] = (bf16_t)v.x; o[1] = (bf16_t)v.y; o[2] = (bf16_t)v.z; o[3] = (bf16_t)v.w;
}

// ---------------------------------------------------------------------------
// GEMM: C[M=8192, N=1024] = A[M,K=1024](bf16) * Wt^T (Wt is [N][K] bf16) + bias
// Block tile 128x64, K-step 64. Tiles staged by the Tensor Data Mover into
// double-buffered LDS (wave 0 issues the DMA, 2 TDM ops kept in flight,
// s_wait_tensorcnt + barrier for consumption). 8 waves, each wave a 32x32
// tile = 2x2 WMMA accumulators, 8 WMMAs per K-step.
// Epilogue: (acc + bias) * scale -> bf16 and/or fp32 output.
// ---------------------------------------------------------------------------
__global__ void __launch_bounds__(256) gemm_kernel(
    const bf16_t* __restrict__ A, const bf16_t* __restrict__ Bt,
    const float* __restrict__ bias, bf16_t* __restrict__ outB,
    float* __restrict__ outF, float scale) {
  __shared__ alignas(16) bf16_t As[2][128][72];
  __shared__ alignas(16) bf16_t Bs[2][64][72];

  const int tid = threadIdx.x;
  const int lane = tid & 31, w = tid >> 5;
  const int wm = w & 3, wn = w >> 2;
  const int rowBase = blockIdx.y * 128;
  const int colBase = blockIdx.x * 64;

  const v8f vz = {0.f, 0.f, 0.f, 0.f, 0.f, 0.f, 0.f, 0.f};
  v8f acc[2][2];
#pragma unroll
  for (int i = 0; i < 2; ++i)
#pragma unroll
    for (int j = 0; j < 2; ++j) acc[i][j] = vz;

  const bf16_t* aBase = A  + (size_t)rowBase * 1024;
  const bf16_t* bBase = Bt + (size_t)colBase * 1024;

  // Prime buffer 0.
  if (w == 0) {
    tdm_load_2d(aBase, lds_off(&As[0][0][0]), 64, 128, 1024, 8192);
    tdm_load_2d(bBase, lds_off(&Bs[0][0][0]), 64, 64, 1024, 1024);
  }

  for (int kt = 0; kt < 16; ++kt) {
    const int cur = kt & 1;
    if (w == 0) {
      if (kt + 1 < 16) {
        // Queue the next K-slice into the other buffer, then wait for the
        // current pair only (TDM ops complete in order within a wave).
        tdm_load_2d(aBase + (kt + 1) * 64, lds_off(&As[cur ^ 1][0][0]),
                    64, 128, 1024, 8192);
        tdm_load_2d(bBase + (kt + 1) * 64, lds_off(&Bs[cur ^ 1][0][0]),
                    64, 64, 1024, 1024);
        __builtin_amdgcn_s_wait_tensorcnt((short)2);
      } else {
        __builtin_amdgcn_s_wait_tensorcnt((short)0);
      }
    }
    __syncthreads();   // current tiles visible to all waves

    const bf16_t* as = &As[cur][0][0];
    const bf16_t* bs = &Bs[cur][0][0];
#pragma unroll
    for (int kk = 0; kk < 64; kk += 32) {
      v16bf a0 = load_frag(as, 72, wm * 32,      kk, lane);
      v16bf a1 = load_frag(as, 72, wm * 32 + 16, kk, lane);
      v16bf b0 = load_frag(bs, 72, wn * 32,      kk, lane);
      v16bf b1 = load_frag(bs, 72, wn * 32 + 16, kk, lane);
      acc[0][0] = WMMA_BF16(a0, b0, acc[0][0]);
      acc[0][1] = WMMA_BF16(a0, b1, acc[0][1]);
      acc[1][0] = WMMA_BF16(a1, b0, acc[1][0]);
      acc[1][1] = WMMA_BF16(a1, b1, acc[1][1]);
    }
    __syncthreads();   // all reads of `cur` done before it is re-targeted
  }

  // Epilogue. C layout: lane&15 = N column; M = vgpr + 8*(lane>>4).
  const int rb = (lane >> 4) << 3;
  const int cb = lane & 15;
#pragma unroll
  for (int fm = 0; fm < 2; ++fm)
#pragma unroll
    for (int fn = 0; fn < 2; ++fn) {
      int col = colBase + wn * 32 + fn * 16 + cb;
      float bv = bias[col];
#pragma unroll
      for (int r = 0; r < 8; ++r) {
        int row = rowBase + wm * 32 + fm * 16 + rb + r;
        float v = (acc[fm][fn][r] + bv) * scale;
        size_t idx = (size_t)row * 1024 + col;
        if (outB) outB[idx] = (bf16_t)v;
        if (outF) outF[idx] = v;
      }
    }
}

// ---------------------------------------------------------------------------
// Flash attention: one block per (batch, head, 64-query tile).
// Q already carries the 1/sqrt(64) scale. Streams 8 key tiles of 64.
// ---------------------------------------------------------------------------
__global__ void __launch_bounds__(256) attn_kernel(
    const bf16_t* __restrict__ Q, const bf16_t* __restrict__ K,
    const bf16_t* __restrict__ V, const float* __restrict__ mask, int moff,
    bf16_t* __restrict__ O) {
  __shared__ alignas(16) bf16_t Qs[64][72];
  __shared__ alignas(16) bf16_t Ks[64][72];
  __shared__ alignas(16) bf16_t VsT[64][72];   // VsT[feature][key]
  __shared__ alignas(16) bf16_t Ps[64][72];
  __shared__ float Ss[64][68];
  __shared__ float rowmax[64], rowsum[64], rowscale[64];

  const int tid = threadIdx.x;
  const int lane = tid & 31, w = tid >> 5;
  const int mT = w & 3, nT = w >> 2;          // wave tile: 16 rows x 32 cols
  const int b = blockIdx.z, h = blockIdx.y, q0 = blockIdx.x * 64;
  const int headc = h * 64;
  const float* mrow = mask + b * 1024 + moff;

  // Stage the 64x64 Q tile.
#pragma unroll
  for (int i = 0; i < 2; ++i) {
    int c = tid + i * 256;
    int r = c >> 3, kc = (c & 7) * 8;
    *(v8bf*)&Qs[r][kc] =
        *(const v8bf*)&Q[(size_t)(b * 512 + q0 + r) * 1024 + headc + kc];
  }
  if (tid < 64) { rowmax[tid] = -1e30f; rowsum[tid] = 0.f; }

  const v8f vz = {0.f, 0.f, 0.f, 0.f, 0.f, 0.f, 0.f, 0.f};
  v8f o0 = vz, o1 = vz;
  __syncthreads();

  for (int kt = 0; kt < 8; ++kt) {
    const int s0 = kt * 64;
    // Stage K tile row-major (= B^T layout for scores) and V transposed.
#pragma unroll
    for (int i = 0; i < 2; ++i) {
      int c = tid + i * 256;
      int r = c >> 3, kc = (c & 7) * 8;
      size_t grow = (size_t)(b * 512 + s0 + r) * 1024 + headc + kc;
      *(v8bf*)&Ks[r][kc] = *(const v8bf*)&K[grow];
      v8bf vv = *(const v8bf*)&V[grow];
#pragma unroll
      for (int j = 0; j < 8; ++j) VsT[kc + j][r] = vv[j];
    }
    __syncthreads();

    // S tile = Q * K^T  (K dim = 64 features -> two WMMA steps)
    v8f s0a = vz, s1a = vz;
#pragma unroll
    for (int kk = 0; kk < 64; kk += 32) {
      v16bf a  = load_frag(&Qs[0][0], 72, mT * 16, kk, lane);
      v16bf bb0 = load_frag(&Ks[0][0], 72, nT * 32,      kk, lane);
      v16bf bb1 = load_frag(&Ks[0][0], 72, nT * 32 + 16, kk, lane);
      s0a = WMMA_BF16(a, bb0, s0a);
      s1a = WMMA_BF16(a, bb1, s1a);
    }
    // Write scores with mask applied (col depends only on lane -> 2 loads).
    {
      int colA = nT * 32 + (lane & 15);
      int colB = colA + 16;
      bool mA = (mrow[s0 + colA] == 0.f);
      bool mB = (mrow[s0 + colB] == 0.f);
      int rbase = mT * 16 + ((lane >> 4) << 3);
#pragma unroll
      for (int r = 0; r < 8; ++r) {
        Ss[rbase + r][colA] = mA ? -1e9f : s0a[r];
        Ss[rbase + r][colB] = mB ? -1e9f : s1a[r];
      }
    }
    __syncthreads();

    // Online softmax: wave w owns rows 8w..8w+7; lane owns cols lane, lane+32.
    for (int rr = 0; rr < 8; ++rr) {
      int row = w * 8 + rr;
      float a0 = Ss[row][lane], a1 = Ss[row][lane + 32];
      float mx = fmaxf(a0, a1);
#pragma unroll
      for (int off2 = 16; off2; off2 >>= 1) mx = fmaxf(mx, __shfl_xor(mx, off2));
      float Mold = rowmax[row];
      float Mnew = fmaxf(Mold, mx);
      float p0 = __expf(a0 - Mnew), p1 = __expf(a1 - Mnew);
      float ps = p0 + p1;
#pragma unroll
      for (int off2 = 16; off2; off2 >>= 1) ps += __shfl_xor(ps, off2);
      Ps[row][lane]      = (bf16_t)p0;
      Ps[row][lane + 32] = (bf16_t)p1;
      if (lane == 0) {
        float sc = __expf(Mold - Mnew);
        rowscale[row] = sc;
        rowsum[row] = rowsum[row] * sc + ps;
        rowmax[row] = Mnew;
      }
    }
    __syncthreads();

    // Rescale running output, then accumulate P * V.
    {
      int rbase = mT * 16 + ((lane >> 4) << 3);
#pragma unroll
      for (int r = 0; r < 8; ++r) {
        float sc = rowscale[rbase + r];
        o0[r] *= sc;
        o1[r] *= sc;
      }
    }
#pragma unroll
    for (int kk = 0; kk < 64; kk += 32) {
      v16bf a  = load_frag(&Ps[0][0], 72, mT * 16, kk, lane);
      v16bf bb0 = load_frag(&VsT[0][0], 72, nT * 32,      kk, lane);
      v16bf bb1 = load_frag(&VsT[0][0], 72, nT * 32 + 16, kk, lane);
      o0 = WMMA_BF16(a, bb0, o0);
      o1 = WMMA_BF16(a, bb1, o1);
    }
    __syncthreads();
  }

  // Normalize and store as bf16 (input to the O-projection GEMM).
  const int rbase = mT * 16 + ((lane >> 4) << 3);
  const int colA = headc + nT * 32 + (lane & 15);
#pragma unroll
  for (int r = 0; r < 8; ++r) {
    int row = rbase + r;
    float inv = 1.f / rowsum[row];
    size_t grow = (size_t)(b * 512 + q0 + row) * 1024;
    O[grow + colA]      = (bf16_t)(o0[r] * inv);
    O[grow + colA + 16] = (bf16_t)(o1[r] * inv);
  }
}

// ---------------------------------------------------------------------------
// Residual add + LayerNorm over D=1024 (one block per row).
// ---------------------------------------------------------------------------
__global__ void __launch_bounds__(256) add_ln_kernel(
    const float* __restrict__ Oproj, const float* __restrict__ emb, int off,
    const float* __restrict__ g, const float* __restrict__ bb,
    float* __restrict__ out) {
  __shared__ float red[8];
  __shared__ float s_mu, s_rstd;
  const int row = blockIdx.x;
  const int b = row >> 9, s = row & 511;
  const float* resid = emb + (((size_t)(b * 1024 + off + s)) << 10);
  const float* x = Oproj + ((size_t)row << 10);

  float v[4];
  float lsum = 0.f;
#pragma unroll
  for (int i = 0; i < 4; ++i) {
    int c = threadIdx.x + i * 256;
    v[i] = x[c] + resid[c];
    lsum += v[i];
  }
#pragma unroll
  for (int o2 = 16; o2; o2 >>= 1) lsum += __shfl_xor(lsum, o2);
  if ((threadIdx.x & 31) == 0) red[threadIdx.x >> 5] = lsum;
  __syncthreads();
  if (threadIdx.x == 0) {
    float t = 0.f;
    for (int i = 0; i < 8; ++i) t += red[i];
    s_mu = t * (1.f / 1024.f);
  }
  __syncthreads();
  const float mu = s_mu;

  float lvar = 0.f;
#pragma unroll
  for (int i = 0; i < 4; ++i) {
    float d = v[i] - mu;
    lvar += d * d;
  }
#pragma unroll
  for (int o2 = 16; o2; o2 >>= 1) lvar += __shfl_xor(lvar, o2);
  __syncthreads();
  if ((threadIdx.x & 31) == 0) red[threadIdx.x >> 5] = lvar;
  __syncthreads();
  if (threadIdx.x == 0) {
    float t = 0.f;
    for (int i = 0; i < 8; ++i) t += red[i];
    s_rstd = rsqrtf(t * (1.f / 1024.f) + 1e-5f);
  }
  __syncthreads();
  const float rstd = s_rstd;
#pragma unroll
  for (int i = 0; i < 4; ++i) {
    int c = threadIdx.x + i * 256;
    out[((size_t)row << 10) + c] = (v[i] - mu) * rstd * g[c] + bb[c];
  }
}

// ---------------------------------------------------------------------------
// Masked mean pooling into feats[b][sec*1024 + f].
// sec: 0=premise(raw), 1=hypothesis(raw), 2=p2h, 3=h2p.
// ---------------------------------------------------------------------------
__global__ void __launch_bounds__(256) pool_kernel(
    const float* __restrict__ emb, const float* __restrict__ p2h,
    const float* __restrict__ h2p, const float* __restrict__ mask,
    float* __restrict__ feats) {
  const int b = blockIdx.x, sec = blockIdx.y, tid = threadIdx.x;
  const float* src;
  int moff;
  if (sec == 0)      { src = emb + (size_t)b * 1048576;              moff = 0;   }
  else if (sec == 1) { src = emb + (size_t)b * 1048576 + 512 * 1024; moff = 512; }
  else if (sec == 2) { src = p2h + (size_t)b * 524288;               moff = 0;   }
  else               { src = h2p + (size_t)b * 524288;               moff = 512; }
  const float* m = mask + b * 1024 + moff;

  float acc[4] = {0.f, 0.f, 0.f, 0.f};
  float dsum = 0.f;
  for (int s = 0; s < 512; ++s) {
    float mv = m[s];
    dsum += mv;
    const float* rp = src + (size_t)s * 1024;
#pragma unroll
    for (int i = 0; i < 4; ++i) acc[i] += rp[tid + 256 * i] * mv;
  }
  float denom = fmaxf(dsum, 1e-9f);
#pragma unroll
  for (int i = 0; i < 4; ++i)
    feats[(size_t)b * 4096 + sec * 1024 + tid + 256 * i] = acc[i] / denom;
}

// ---------------------------------------------------------------------------
// Tiny MLP head (<<1% of FLOPs): plain fp32 dot products.
// ---------------------------------------------------------------------------
__global__ void __launch_bounds__(256) mlp1_kernel(
    const float* __restrict__ feats, const float* __restrict__ W1,
    const float* __restrict__ b1, float* __restrict__ h1) {
  int idx = blockIdx.x * 256 + threadIdx.x;   // 16*1024
  int b = idx >> 10, n = idx & 1023;
  const float* f = feats + (size_t)b * 4096;
  float s = b1[n];
  for (int k = 0; k < 4096; ++k) s = fmaf(f[k], W1[(size_t)k * 1024 + n], s);
  h1[idx] = fmaxf(s, 0.f);
}

__global__ void __launch_bounds__(256) mlp2_kernel(
    const float* __restrict__ h1, const float* __restrict__ W2,
    const float* __restrict__ b2, float* __restrict__ h2) {
  int idx = blockIdx.x * 256 + threadIdx.x;   // 16*512
  int b = idx >> 9, n = idx & 511;
  const float* f = h1 + (size_t)b * 1024;
  float s = b2[n];
  for (int k = 0; k < 1024; ++k) s = fmaf(f[k], W2[(size_t)k * 512 + n], s);
  h2[idx] = fmaxf(s, 0.f);
}

__global__ void mlp3_kernel(const float* __restrict__ h2,
                            const float* __restrict__ W3,
                            const float* __restrict__ b3,
                            float* __restrict__ out) {
  int idx = threadIdx.x;
  if (idx >= 48) return;
  int b = idx / 3, n = idx % 3;
  const float* f = h2 + (size_t)b * 512;
  float s = b3[n];
  for (int k = 0; k < 512; ++k) s = fmaf(f[k], W3[k * 3 + n], s);
  out[idx] = s;
}

// ---------------------------------------------------------------------------
// Host driver
// ---------------------------------------------------------------------------
extern "C" void kernel_launch(void* const* d_in, const int* in_sizes, int n_in,
                              void* d_out, int out_size, void* d_ws, size_t ws_size,
                              hipStream_t stream) {
  (void)in_sizes; (void)n_in; (void)out_size; (void)ws_size;
  const float* emb  = (const float*)d_in[0];
  const float* mask = (const float*)d_in[1];
  const float* W[8];
  const float* bvec[8];
  for (int p = 0; p < 2; ++p)
    for (int i = 0; i < 4; ++i) {
      W[p * 4 + i]    = (const float*)d_in[2 + p * 10 + i * 2];
      bvec[p * 4 + i] = (const float*)d_in[3 + p * 10 + i * 2];
    }
  const float* lng[2] = {(const float*)d_in[10], (const float*)d_in[20]};
  const float* lnb[2] = {(const float*)d_in[11], (const float*)d_in[21]};
  const float* W1 = (const float*)d_in[22]; const float* b1 = (const float*)d_in[23];
  const float* W2 = (const float*)d_in[24]; const float* b2 = (const float*)d_in[25];
  const float* W3 = (const float*)d_in[26]; const float* b3 = (const float*)d_in[27];
  float* outp = (float*)d_out;

  char* ws = (char*)d_ws;
  size_t off = 0;
  auto alloc = [&](size_t bytes) -> void* {
    void* p = ws + off;
    off += (bytes + 255) & ~(size_t)255;
    return p;
  };
  bf16_t* Wt[8];
  for (int i = 0; i < 8; ++i) Wt[i] = (bf16_t*)alloc((size_t)1024 * 1024 * 2);
  bf16_t* Xp = (bf16_t*)alloc((size_t)8192 * 1024 * 2);
  bf16_t* Xh = (bf16_t*)alloc((size_t)8192 * 1024 * 2);
  bf16_t* Qb = (bf16_t*)alloc((size_t)8192 * 1024 * 2);
  bf16_t* Kb = (bf16_t*)alloc((size_t)8192 * 1024 * 2);
  bf16_t* Vb = (bf16_t*)alloc((size_t)8192 * 1024 * 2);
  bf16_t* Ab = (bf16_t*)alloc((size_t)8192 * 1024 * 2);
  float*  Of = (float*)alloc((size_t)8192 * 1024 * 4);
  float*  ctx[2];
  ctx[0] = (float*)alloc((size_t)8192 * 1024 * 4);
  ctx[1] = (float*)alloc((size_t)8192 * 1024 * 4);
  float* feats = (float*)alloc((size_t)16 * 4096 * 4);
  float* h1 = (float*)alloc((size_t)16 * 1024 * 4);
  float* h2 = (float*)alloc((size_t)16 * 512 * 4);

  for (int i = 0; i < 8; ++i)
    transpose_convert_kernel<<<dim3(32, 32), dim3(32, 8), 0, stream>>>(W[i], Wt[i]);
  split_bf16_kernel<<<8192, 256, 0, stream>>>(emb, Xp, 0);
  split_bf16_kernel<<<8192, 256, 0, stream>>>(emb, Xh, 512);

  for (int pass = 0; pass < 2; ++pass) {
    const bf16_t* qsrc = pass ? Xh : Xp;
    const bf16_t* ksrc = pass ? Xp : Xh;
    // Q projection folds in 1/sqrt(head_dim) = 0.125 (exact in bf16).
    gemm_kernel<<<dim3(16, 64), 256, 0, stream>>>(qsrc, Wt[pass * 4 + 0],
        bvec[pass * 4 + 0], Qb, (float*)nullptr, 0.125f);
    gemm_kernel<<<dim3(16, 64), 256, 0, stream>>>(ksrc, Wt[pass * 4 + 1],
        bvec[pass * 4 + 1], Kb, (float*)nullptr, 1.0f);
    gemm_kernel<<<dim3(16, 64), 256, 0, stream>>>(ksrc, Wt[pass * 4 + 2],
        bvec[pass * 4 + 2], Vb, (float*)nullptr, 1.0f);
    attn_kernel<<<dim3(8, 16, 16), 256, 0, stream>>>(Qb, Kb, Vb, mask,
        pass ? 0 : 512, Ab);
    gemm_kernel<<<dim3(16, 64), 256, 0, stream>>>(Ab, Wt[pass * 4 + 3],
        bvec[pass * 4 + 3], (bf16_t*)nullptr, Of, 1.0f);
    add_ln_kernel<<<8192, 256, 0, stream>>>(Of, emb, pass ? 512 : 0,
        lng[pass], lnb[pass], ctx[pass]);
  }

  pool_kernel<<<dim3(16, 4), 256, 0, stream>>>(emb, ctx[0], ctx[1], mask, feats);
  mlp1_kernel<<<64, 256, 0, stream>>>(feats, W1, b1, h1);
  mlp2_kernel<<<32, 256, 0, stream>>>(h1, W2, b2, h2);
  mlp3_kernel<<<1, 64, 0, stream>>>(h2, W3, b3, outp);
}
